// VectorQuantizer_83811991814255
// MI455X (gfx1250) — matrix operand order repaired and checked by reading, not compiled, not run on hardware
//
#include <hip/hip_runtime.h>

typedef __bf16 bf16_t;
typedef __attribute__((ext_vector_type(16))) __bf16 v16bf;
typedef __attribute__((ext_vector_type(8)))  __bf16 v8bf;
typedef __attribute__((ext_vector_type(8)))  float  v8f;

#define NUM_CODES 8192
#define CODE_DIM  256
#define NTOK      32768            // 8*64*64
#define TILE_ELEMS (16 * CODE_DIM) // one 16-code tile = 4096 bf16

// ---------------------------------------------------------------------------
// Fragment loaders per CDNA5 WMMA VGPR layouts (wave32).
// A 16x32 bf16: lane holds row M=lane&15; K chunks at kb and kb+16, kb=(lane>>4)*8
__device__ __forceinline__ v16bf load_a_frag(const bf16_t* __restrict__ p) {
  v8bf lo = *(const v8bf*)(p);
  v8bf hi = *(const v8bf*)(p + 16);
  return __builtin_shufflevector(lo, hi, 0,1,2,3,4,5,6,7,8,9,10,11,12,13,14,15);
}
// B 32x16 bf16: lane holds column N=lane&15; K chunks at kr and kr+16, kr=(lane>>4)*8
__device__ __forceinline__ v16bf load_b_frag(const bf16_t* p) {
  v8bf lo = *(const v8bf*)(p);
  v8bf hi = *(const v8bf*)(p + 16);
  return __builtin_shufflevector(lo, hi, 0,1,2,3,4,5,6,7,8,9,10,11,12,13,14,15);
}

// ---------------------------------------------------------------------------
// Kernel 1: qc_f32[k][d] = dot(codebook[k,:], proj_w[d,:]) + proj_b[d]
__global__ __launch_bounds__(256) void vq_proj_kernel(
    const float* __restrict__ codebook, const float* __restrict__ proj_w,
    const float* __restrict__ proj_b, float* __restrict__ qc) {
  __shared__ float cb[32][CODE_DIM + 1];
  const int kbase = blockIdx.x * 32;
  const int tid = threadIdx.x;
  for (int i = tid; i < 32 * CODE_DIM; i += 256) {
    int c = i >> 8, j = i & 255;
    cb[c][j] = codebook[(long)(kbase + c) * CODE_DIM + j];
  }
  __syncthreads();
  const int d = tid;
  float acc[32];
#pragma unroll
  for (int c = 0; c < 32; ++c) acc[c] = 0.f;
  const float* wrow = proj_w + (long)d * CODE_DIM;
  for (int j = 0; j < CODE_DIM; ++j) {
    float w = wrow[j];
#pragma unroll
    for (int c = 0; c < 32; ++c) acc[c] = fmaf(cb[c][j], w, acc[c]);
  }
  const float b = proj_b[d];
#pragma unroll
  for (int c = 0; c < 32; ++c)
    qc[(long)(kbase + c) * CODE_DIM + d] = acc[c] + b;
}

// ---------------------------------------------------------------------------
// Kernel 2: per-code row norm + hi/lo bf16 split of projected codebook.
__global__ __launch_bounds__(256) void vq_prep_codes_kernel(
    const float* __restrict__ qc, bf16_t* __restrict__ qh,
    bf16_t* __restrict__ ql, float* __restrict__ c2) {
  const int k = blockIdx.x, t = threadIdx.x;
  float v = qc[(long)k * CODE_DIM + t];
  bf16_t h = (bf16_t)v;
  float hf = (float)h;
  bf16_t l = (bf16_t)(v - hf);
  qh[(long)k * CODE_DIM + t] = h;
  ql[(long)k * CODE_DIM + t] = l;
  float s = v * v;
#pragma unroll
  for (int off = 16; off > 0; off >>= 1) s += __shfl_xor(s, off, 32);
  __shared__ float red[8];
  if ((t & 31) == 0) red[t >> 5] = s;
  __syncthreads();
  if (t == 0) {
    float tot = 0.f;
#pragma unroll
    for (int w = 0; w < 8; ++w) tot += red[w];
    c2[k] = tot;
  }
}

// ---------------------------------------------------------------------------
// Kernel 3: hi/lo bf16 split of z (vectorized x4).
__global__ __launch_bounds__(256) void vq_split_z_kernel(
    const float* __restrict__ z, bf16_t* __restrict__ zh, bf16_t* __restrict__ zl) {
  long i = ((long)blockIdx.x * 256 + threadIdx.x) * 4;
  float4 v = *(const float4*)(z + i);
  float vv[4] = {v.x, v.y, v.z, v.w};
#pragma unroll
  for (int j = 0; j < 4; ++j) {
    bf16_t h = (bf16_t)vv[j];
    zh[i + j] = h;
    zl[i + j] = (bf16_t)(vv[j] - (float)h);
  }
}

// ---------------------------------------------------------------------------
// Kernel 4: fused distance-GEMM + argmin, LDS double-buffered B tiles.
// 8 waves/block, each wave owns one 16-row M tile; A frags (K=256, hi+lo)
// resident in VGPRs. Codebook tiles (16 codes, hi+lo = 16KB) are staged
// cooperatively into LDS, double-buffered: global loads for tile nt+1 are
// issued into registers before the 24 WMMAs of tile nt, stores + one barrier
// after. score = c2[n] - 2*z.c  (||z||^2 constant per row).
__global__ __launch_bounds__(256) void vq_argmin_kernel(
    const bf16_t* __restrict__ zhi, const bf16_t* __restrict__ zlo,
    const bf16_t* __restrict__ chi, const bf16_t* __restrict__ clo,
    const float* __restrict__ c2, int* __restrict__ idx_out) {
  __shared__ bf16_t lds_tile[2][2 * TILE_ELEMS];  // [buf][hi | lo], 16KB each

  const int tid = threadIdx.x;
  const int lane = tid & 31;
  const int wave = tid >> 5;
  const int mtile = blockIdx.x * 8 + wave;
  const int arow = mtile * 16 + (lane & 15);
  const int kb = (lane >> 4) * 8;     // A-frag K sub-offset
  const int kr = (lane >> 4) * 8;     // B-frag K sub-offset
  const int ncol = lane & 15;         // this lane's N column in every tile

  // Resident A fragments (hi + lo): 16 x 256 bf16 per wave.
  v16bf ahi[8], alo[8];
  const bf16_t* pzh = zhi + (long)arow * CODE_DIM + kb;
  const bf16_t* pzl = zlo + (long)arow * CODE_DIM + kb;
#pragma unroll
  for (int kk = 0; kk < 8; ++kk) {
    ahi[kk] = load_a_frag(pzh + kk * 32);
    alo[kk] = load_a_frag(pzl + kk * 32);
  }

  // Staging: threads 0..127 stage hi half, 128..255 stage lo half; 64B each.
  const bf16_t* gsrc = (tid < 128) ? chi : clo;
  const int toff = (tid & 127) * 32;                       // element offset in tile
  const int lhalf = (tid < 128) ? 0 : TILE_ELEMS;
  bf16_t* lst0 = &lds_tile[0][lhalf + toff];
  bf16_t* lst1 = &lds_tile[1][lhalf + toff];

  uint4 st0, st1, st2, st3;
  {  // prologue: tile 0 -> buf 0
    const uint4* p = (const uint4*)(gsrc + toff);
    st0 = p[0]; st1 = p[1]; st2 = p[2]; st3 = p[3];
    uint4* q = (uint4*)lst0;
    q[0] = st0; q[1] = st1; q[2] = st2; q[3] = st3;
  }
  __syncthreads();

  float best[8];
  int bidx[8];
#pragma unroll
  for (int r = 0; r < 8; ++r) { best[r] = 3.4e38f; bidx[r] = 0; }

  for (int nt = 0; nt < NUM_CODES / 16; ++nt) {
    // Issue global loads for tile nt+1 early (latency hidden by WMMAs below).
    if (nt + 1 < NUM_CODES / 16) {
      const uint4* p = (const uint4*)(gsrc + (long)(nt + 1) * TILE_ELEMS + toff);
      st0 = p[0]; st1 = p[1]; st2 = p[2]; st3 = p[3];
      __builtin_prefetch(gsrc + (long)(nt + 2) * TILE_ELEMS + toff, 0, 1);
    }
    const float c2v = c2[nt * 16 + ncol];

    // Consume tile nt from LDS: one DS clause of fragments, then WMMA chain.
    const bf16_t* tb = &lds_tile[nt & 1][ncol * CODE_DIM + kr];
    v16bf bh[8], bl[8];
#pragma unroll
    for (int kk = 0; kk < 8; ++kk) {
      bh[kk] = load_b_frag(tb + kk * 32);
      bl[kk] = load_b_frag(tb + TILE_ELEMS + kk * 32);
    }
    v8f acc = {};
#pragma unroll
    for (int kk = 0; kk < 8; ++kk) {
      acc = __builtin_amdgcn_wmma_f32_16x16x32_bf16(false, ahi[kk], false, bh[kk], (short)0, acc, false, false);
      acc = __builtin_amdgcn_wmma_f32_16x16x32_bf16(false, alo[kk], false, bh[kk], (short)0, acc, false, false);
      acc = __builtin_amdgcn_wmma_f32_16x16x32_bf16(false, ahi[kk], false, bl[kk], (short)0, acc, false, false);
    }
#pragma unroll
    for (int r = 0; r < 8; ++r) {
      float s = fmaf(-2.0f, acc[r], c2v);
      if (s < best[r]) { best[r] = s; bidx[r] = nt * 16 + ncol; }
    }

    // Store tile nt+1 into the other buffer; one barrier per tile.
    if (nt + 1 < NUM_CODES / 16) {
      uint4* q = (uint4*)((nt & 1) ? lst0 : lst1);
      q[0] = st0; q[1] = st1; q[2] = st2; q[3] = st3;
    }
    __syncthreads();
  }

  // D layout: lane holds N=lane&15; register r -> row M = r + 8*(lane>>4).
#pragma unroll
  for (int m = 8; m >= 1; m >>= 1) {
#pragma unroll
    for (int r = 0; r < 8; ++r) {
      float ov = __shfl_xor(best[r], m, 32);
      int oi = __shfl_xor(bidx[r], m, 32);
      if (ov < best[r] || (ov == best[r] && oi < bidx[r])) { best[r] = ov; bidx[r] = oi; }
    }
  }
  if ((lane & 15) == 0) {
    int base = mtile * 16 + (lane >> 4) * 8;
#pragma unroll
    for (int r = 0; r < 8; ++r) idx_out[base + r] = bidx[r];
  }
}

// ---------------------------------------------------------------------------
// Kernel 5: z_q gather, float4, 4 rows per block.
__global__ __launch_bounds__(256) void vq_gather_kernel(
    const float* __restrict__ qc, const int* __restrict__ idx, float* __restrict__ out) {
  const int row = blockIdx.x * 4 + (threadIdx.x >> 6);
  const int c = (threadIdx.x & 63) * 4;
  const int k = idx[row];
  float4 v = *(const float4*)(qc + (long)k * CODE_DIM + c);
  *(float4*)(out + (long)row * CODE_DIM + c) = v;
}

// ---------------------------------------------------------------------------
extern "C" void kernel_launch(void* const* d_in, const int* in_sizes, int n_in,
                              void* d_out, int out_size, void* d_ws, size_t ws_size,
                              hipStream_t stream) {
  const float* z        = (const float*)d_in[0];
  const float* codebook = (const float*)d_in[1];
  const float* proj_w   = (const float*)d_in[2];
  const float* proj_b   = (const float*)d_in[3];

  float* out = (float*)d_out;
  int* idx_out = (int*)(out + (long)NTOK * CODE_DIM);   // indices after z_q

  char* ws = (char*)d_ws;
  float*  qc_f32 = (float*)(ws);                          //  8 MB
  bf16_t* qc_hi  = (bf16_t*)(ws + (8ull  << 20));         //  4 MB
  bf16_t* qc_lo  = (bf16_t*)(ws + (12ull << 20));         //  4 MB
  float*  c2     = (float*)(ws + (16ull << 20));          // 32 KB
  bf16_t* z_hi   = (bf16_t*)(ws + (17ull << 20));         // 16 MB
  bf16_t* z_lo   = (bf16_t*)(ws + (33ull << 20));         // 16 MB

  vq_proj_kernel<<<NUM_CODES / 32, 256, 0, stream>>>(codebook, proj_w, proj_b, qc_f32);
  vq_prep_codes_kernel<<<NUM_CODES, 256, 0, stream>>>(qc_f32, qc_hi, qc_lo, c2);
  vq_split_z_kernel<<<(NTOK * CODE_DIM) / (256 * 4), 256, 0, stream>>>(z, z_hi, z_lo);
  vq_argmin_kernel<<<NTOK / (16 * 8), 256, 0, stream>>>(z_hi, z_lo, qc_hi, qc_lo, c2, idx_out);
  vq_gather_kernel<<<NTOK / 4, 256, 0, stream>>>(qc_f32, idx_out, out);
}
